// NodeNetwork_49082886258999
// MI455X (gfx1250) — matrix operand order repaired and checked by reading, not compiled, not run on hardware
//
#include <hip/hip_runtime.h>

// ---------------------------------------------------------------------------
// Problem constants (from reference setup_inputs)
// ---------------------------------------------------------------------------
#define BB 4
#define NN 2048
#define EE 8192
#define DD 64

typedef __attribute__((ext_vector_type(16))) __bf16          v16bf;
typedef __attribute__((ext_vector_type(8)))  float           v8f;
typedef __attribute__((ext_vector_type(2)))  float           v2f;
typedef __attribute__((ext_vector_type(8)))  unsigned short  v8us;

// f32 -> bf16 bits, round-to-nearest-even (used off the hot path)
__device__ __forceinline__ unsigned short f2bf(float x) {
    unsigned u = __builtin_bit_cast(unsigned, x);
    u += 0x7fffu + ((u >> 16) & 1u);
    return (unsigned short)(u >> 16);
}
// Fast truncating pack of two f32 high-halves -> one dword (hot fill paths).
// ~1 VALU (v_perm) vs ~7 for RNE; adds <=0.5 ulp(bf16) on streaming operands.
__device__ __forceinline__ unsigned pack2(float a, float b) {
    unsigned ua = __builtin_bit_cast(unsigned, a);
    unsigned ub = __builtin_bit_cast(unsigned, b);
    return (ua >> 16) | (ub & 0xffff0000u);
}

// ---------------------------------------------------------------------------
// Fragment loaders. LDS tiles stored k-contiguous: each lane's fragment is two
// contiguous 16B runs -> ds_load_b128 pairs, conflict-free strides (80/144/400B).
//  A 16x32 bf16 (layout [m][k]): lanes 0-15 M=lane  K={0..7,16..23}
//                                lanes16-31 M=l-16  K={8..15,24..31}
//  B 32x16 bf16 (layout [n][k]): lanes 0-15 N=lane  K={0..7,16..23}
//                                lanes16-31 N=l-16  K={8..15,24..31}
// ---------------------------------------------------------------------------
__device__ __forceinline__ v16bf ld_a_frag(const unsigned short* tile,
                                           int stride, int lane) {
    const int m  = lane & 15;
    const int kb = (lane < 16) ? 0 : 8;
    const unsigned short* p = tile + m * stride + kb;
    union { v16bf v; v8us h[2]; } u;
    u.h[0] = *(const v8us*)(p);        // k = kb .. kb+7
    u.h[1] = *(const v8us*)(p + 16);   // k = kb+16 .. kb+23
    return u.v;
}
__device__ __forceinline__ v16bf ld_b_frag(const unsigned short* tile,
                                           int stride, int lane) {
    const int n  = lane & 15;
    const int kb = (lane < 16) ? 0 : 8;
    const unsigned short* p = tile + n * stride + kb;
    union { v16bf v; v8us h[2]; } u;
    u.h[0] = *(const v8us*)(p);
    u.h[1] = *(const v8us*)(p + 16);
    return u.v;
}
__device__ __forceinline__ v8f wmma_bf16(v16bf a, v16bf b, v8f c) {
    return __builtin_amdgcn_wmma_f32_16x16x32_bf16(false, a, false, b,
                                                   (short)0, c, false, false);
}

// ---------------------------------------------------------------------------
// Kernel X: one-time pack/transpose X[b][n][d] f32 -> XbfT[b][d][n] bf16 (1MB)
// so kernel A's X-tiles become contiguous b128 loads. Grid (N/64, B).
// ---------------------------------------------------------------------------
__global__ __launch_bounds__(256) void k_xpack(const float* __restrict__ X,
                                               unsigned short* __restrict__ XbfT) {
    const int tid = threadIdx.x;
    const int b   = blockIdx.y;
    const int n0  = blockIdx.x * 64;
    const int d   = tid & 63;
#pragma unroll
    for (int cc = 0; cc < 2; ++cc) {
        int  nb = n0 + ((tid >> 6) * 2 + cc) * 8;
        v8us v;
#pragma unroll
        for (int i = 0; i < 8; ++i)
            v[i] = f2bf(X[(long)(b * NN + nb + i) * DD + d]);
        *(v8us*)&XbfT[(long)(b * DD + d) * NN + nb] = v;
    }
}

// ---------------------------------------------------------------------------
// Kernel A: bo = Ro^T X , bi = Ri^T X. Grid (E/64, B), 8 waves/block.
// Software-pipelined: regs <- global(next), wmma(cur), LDS(next) <- regs.
// Results stored TRANSPOSED [b][d][e] bf16 so kernel B's tile loads are
// contiguous b128 along e.
// ---------------------------------------------------------------------------
__global__ __launch_bounds__(256) void k_bobi(
    const unsigned short* __restrict__ XbfT, const float* __restrict__ Ri,
    const float* __restrict__ Ro, unsigned short* __restrict__ boT,
    unsigned short* __restrict__ biT) {
    __shared__ __align__(16) unsigned short Xf[2][64 * 40];   // [d][k=n]
    __shared__ __align__(16) unsigned short Rof[2][64 * 40];  // [e][k=n]
    __shared__ __align__(16) unsigned short Rif[2][64 * 40];

    const int tid   = threadIdx.x;
    const int lane  = tid & 31;
    const int wave  = tid >> 5;
    const int b     = blockIdx.y;
    const int e0    = blockIdx.x * 64;
    const int strip = wave & 3;

    v8f acc[4];
#pragma unroll
    for (int t = 0; t < 4; ++t)
#pragma unroll
        for (int i = 0; i < 8; ++i) acc[t][i] = 0.0f;

    // per-thread fill geometry
    const int  col   = tid & 63;                              // R column (e)
    const int  rp0   = 2 * (tid >> 6);                        // R row-pairs
    const int  dX    = tid >> 2;                              // X d-row
    const int  nch   = (tid & 3) * 8;                         // X k-chunk
    const long Rbase = (long)(b * NN + rp0) * EE + e0 + col;  // + (n0+8j)*EE
    const long Xtb   = (long)(b * DD + dX) * NN + nch;        // + n0

    float rro[8], rri[8];
    v8us  ux;

#define KA_LOAD(n0_)                                           \
    ux = *(const v8us*)&XbfT[Xtb + (n0_)];                     \
    _Pragma("unroll") for (int j = 0; j < 4; ++j) {            \
        long ofE = (long)((n0_) + 8 * j) * EE;                 \
        rro[2 * j]     = Ro[Rbase + ofE];                      \
        rro[2 * j + 1] = Ro[Rbase + ofE + EE];                 \
        rri[2 * j]     = Ri[Rbase + ofE];                      \
        rri[2 * j + 1] = Ri[Rbase + ofE + EE];                 \
    }
#define KA_STORE(buf_)                                                       \
    *(v8us*)&Xf[buf_][dX * 40 + nch] = ux;                                   \
    _Pragma("unroll") for (int j = 0; j < 4; ++j) {                          \
        int r0 = rp0 + 8 * j;                                                \
        *(unsigned*)&Rof[buf_][col * 40 + r0] = pack2(rro[2 * j], rro[2 * j + 1]); \
        *(unsigned*)&Rif[buf_][col * 40 + r0] = pack2(rri[2 * j], rri[2 * j + 1]); \
    }

    KA_LOAD(0);
    KA_STORE(0);
    __syncthreads();

    int cur = 0;
    for (int it = 0; it < NN / 32; ++it) {
        const bool hasNext = (it + 1 < NN / 32);
        if (hasNext) {
            KA_LOAD(32 * (it + 1));
            long pf = Rbase + (long)(32 * (it + 2)) * EE;
            __builtin_prefetch(&Ro[pf], 0, 1);
            __builtin_prefetch(&Ri[pf], 0, 1);
        }
        const unsigned short* Rsh = (wave < 4) ? Rof[cur] : Rif[cur];
        v16bf a = ld_a_frag(Rsh + (strip * 16) * 40, 40, lane);
        v16bf bx[4];
#pragma unroll
        for (int t = 0; t < 4; ++t)
            bx[t] = ld_b_frag(Xf[cur] + (t * 16) * 40, 40, lane);
#pragma unroll
        for (int t = 0; t < 4; ++t) acc[t] = wmma_bf16(a, bx[t], acc[t]);
        if (hasNext) KA_STORE(cur ^ 1);
        __syncthreads();
        cur ^= 1;
    }

    // Write transposed: dst[b][d][e]
    unsigned short* dst = (wave < 4) ? boT : biT;
#pragma unroll
    for (int r = 0; r < 8; ++r) {
        int rowm = r + ((lane < 16) ? 0 : 8);
        int e    = e0 + strip * 16 + rowm;
#pragma unroll
        for (int t = 0; t < 4; ++t) {
            int d = t * 16 + (lane & 15);
            dst[(long)(b * DD + d) * EE + e] = f2bf(acc[t][r]);
        }
    }
#undef KA_LOAD
#undef KA_STORE
}

// ---------------------------------------------------------------------------
// Fused MLP layer: z = A(16xK strip) @ W + bias ; LayerNorm(64) ; tanh
// Weights pre-staged in LDS in B-fragment layout [n][k].
// ---------------------------------------------------------------------------
__device__ __forceinline__ void mlp_layer(
    const unsigned short* hsrc, int srcStride, int K,
    const unsigned short* Wf, int wStride, const float* __restrict__ bias,
    const float* __restrict__ g, const float* __restrict__ bb,
    unsigned short* hdst, int dstStride, float* outp, int lane, int strip) {
    v8f z[4];
#pragma unroll
    for (int t = 0; t < 4; ++t)
#pragma unroll
        for (int i = 0; i < 8; ++i) z[t][i] = 0.0f;

    for (int k0 = 0; k0 < K; k0 += 32) {
        v16bf a = ld_a_frag(hsrc + (strip * 16) * srcStride + k0, srcStride, lane);
        v16bf wb[4];
#pragma unroll
        for (int t = 0; t < 4; ++t)
            wb[t] = ld_b_frag(Wf + (t * 16) * wStride + k0, wStride, lane);
#pragma unroll
        for (int t = 0; t < 4; ++t) z[t] = wmma_bf16(a, wb[t], z[t]);
    }

    float bsv[4], gsv[4], bbv[4];
#pragma unroll
    for (int t = 0; t < 4; ++t) {
        int col = t * 16 + (lane & 15);
        bsv[t] = bias[col];
        gsv[t] = g[col];
        bbv[t] = bb[col];
    }

#pragma unroll
    for (int r = 0; r < 8; ++r) {
        float xv[4];
        float s = 0.0f, q = 0.0f;
#pragma unroll
        for (int t = 0; t < 4; ++t) {
            xv[t] = z[t][r] + bsv[t];
            s += xv[t];
            q += xv[t] * xv[t];
        }
#pragma unroll
        for (int off = 1; off < 16; off <<= 1) {  // stays in 16-lane row group
            s += __shfl_xor(s, off, 32);
            q += __shfl_xor(q, off, 32);
        }
        float mean = s * (1.0f / 64.0f);
        float var  = q * (1.0f / 64.0f) - mean * mean;
        float rinv = rsqrtf(var + 1e-5f);
        int   rowm = r + ((lane < 16) ? 0 : 8);
#pragma unroll
        for (int t = 0; t < 4; ++t) {
            float y = tanhf((xv[t] - mean) * rinv * gsv[t] + bbv[t]);
            int   col = t * 16 + (lane & 15);
            if (outp)
                outp[(long)(strip * 16 + rowm) * DD + col] = y;
            else
                hdst[(strip * 16 + rowm) * dstStride + col] = f2bf(y);
        }
    }
}

// Stage a global f32 weight matrix W[K][64] into LDS [n][k] bf16 layout.
__device__ __forceinline__ void stage_w(const float* __restrict__ W, int K,
                                        unsigned short* Wf, int stride,
                                        int tid) {
    const int n = tid & 63;
    for (int kp = tid >> 6; kp < (K >> 1); kp += 4) {
        int k = 2 * kp;
        *(unsigned*)&Wf[n * stride + k] =
            pack2(W[k * DD + n], W[(k + 1) * DD + n]);
    }
}

// ---------------------------------------------------------------------------
// Kernel B: mi = (Ri.e) bo , mo = (Ro.e) bi  then fused 4-layer LN-tanh MLP.
// Grid (N/64, B), 8 waves/block; software-pipelined double-buffered K loop.
// bo/bi arrive transposed [b][d][e] -> one b128 load + one b128 LDS store
// per matrix per thread per tile.
// ---------------------------------------------------------------------------
__global__ __launch_bounds__(256) void k_mimo_mlp(
    const float* __restrict__ X, const float* __restrict__ ev,
    const float* __restrict__ Ri, const float* __restrict__ Ro,
    const unsigned short* __restrict__ boT, const unsigned short* __restrict__ biT,
    const float* __restrict__ W0, const float* __restrict__ b0,
    const float* __restrict__ g0, const float* __restrict__ bb0,
    const float* __restrict__ W1, const float* __restrict__ b1,
    const float* __restrict__ g1, const float* __restrict__ bb1,
    const float* __restrict__ W2, const float* __restrict__ b2,
    const float* __restrict__ g2, const float* __restrict__ bb2,
    const float* __restrict__ W3, const float* __restrict__ b3,
    const float* __restrict__ g3, const float* __restrict__ bb3,
    float* __restrict__ out) {
    __shared__ __align__(16) unsigned short Ai[2][64 * 40];   // (Ri*e)[n][k=e]
    __shared__ __align__(16) unsigned short Ao[2][64 * 40];   // (Ro*e)[n][k=e]
    __shared__ __align__(16) unsigned short Bof[2][64 * 40];  // bo   [d][k=e]
    __shared__ __align__(16) unsigned short Bif[2][64 * 40];  // bi   [d][k=e]
    __shared__ __align__(16) unsigned short h0[64 * 200];     // [row][mi|mo|X]
    __shared__ __align__(16) unsigned short h1[64 * 72];      // MLP activations
    __shared__ __align__(16) unsigned short Wf0[64 * 200];    // W0 [n][k]
    __shared__ __align__(16) unsigned short Wf1[64 * 72];
    __shared__ __align__(16) unsigned short Wf2[64 * 72];
    __shared__ __align__(16) unsigned short Wf3[64 * 72];

    const int tid   = threadIdx.x;
    const int lane  = tid & 31;
    const int wave  = tid >> 5;
    const int b     = blockIdx.y;
    const int n0    = blockIdx.x * 64;
    const int strip = wave & 3;

    // One-time weight staging (covered by the prologue barrier below).
    stage_w(W0, 192, Wf0, 200, tid);
    stage_w(W1, 64, Wf1, 72, tid);
    stage_w(W2, 64, Wf2, 72, tid);
    stage_w(W3, 64, Wf3, 72, tid);

    v8f acc[4];
#pragma unroll
    for (int t = 0; t < 4; ++t)
#pragma unroll
        for (int i = 0; i < 8; ++i) acc[t][i] = 0.0f;

    // fill geometry
    const int  rowA  = tid >> 4;         // A-row 0..15 (+16j)
    const int  cA0   = (tid & 15) * 2;   // A k-pair base (even)
    const int  dcol  = tid >> 2;         // B d-row 0..63
    const int  el0   = (tid & 3) * 8;    // B k-chunk base (8 halves)
    const long Abase = (long)(b * NN + n0 + rowA) * EE + cA0;  // + e0k + j*16*EE
    const long Bbase = (long)(b * DD + dcol) * EE + el0;       // + e0k
    const long evb   = (long)b * EE + cA0;                     // + e0k

    v2f  fri[4], fro[4], esv;
    v8us ubo, ubi;

#define KB_LOAD(e0k_)                                          \
    esv = *(const v2f*)&ev[evb + (e0k_)];                      \
    _Pragma("unroll") for (int j = 0; j < 4; ++j) {            \
        long gA = Abase + (e0k_) + (long)j * 16 * EE;          \
        fri[j] = *(const v2f*)&Ri[gA];                         \
        fro[j] = *(const v2f*)&Ro[gA];                         \
    }                                                          \
    ubo = *(const v8us*)&boT[Bbase + (e0k_)];                  \
    ubi = *(const v8us*)&biT[Bbase + (e0k_)];
#define KB_STORE(buf_)                                                        \
    _Pragma("unroll") for (int j = 0; j < 4; ++j) {                           \
        int row = rowA + 16 * j;                                              \
        *(unsigned*)&Ai[buf_][row * 40 + cA0] =                               \
            pack2(fri[j][0] * esv[0], fri[j][1] * esv[1]);                    \
        *(unsigned*)&Ao[buf_][row * 40 + cA0] =                               \
            pack2(fro[j][0] * esv[0], fro[j][1] * esv[1]);                    \
    }                                                                         \
    *(v8us*)&Bof[buf_][dcol * 40 + el0] = ubo;                                \
    *(v8us*)&Bif[buf_][dcol * 40 + el0] = ubi;

    KB_LOAD(0);
    KB_STORE(0);
    __syncthreads();

    int cur = 0;
    for (int it = 0; it < EE / 32; ++it) {
        const bool hasNext = (it + 1 < EE / 32);
        if (hasNext) {
            KB_LOAD(32 * (it + 1));
            long pf = Abase + 32 * (it + 2);
            __builtin_prefetch(&Ri[pf], 0, 1);
            __builtin_prefetch(&Ro[pf], 0, 1);
        }
        const unsigned short* Ash = (wave < 4) ? Ai[cur] : Ao[cur];
        const unsigned short* Bsh = (wave < 4) ? Bof[cur] : Bif[cur];
        v16bf a = ld_a_frag(Ash + (strip * 16) * 40, 40, lane);
        v16bf bx[4];
#pragma unroll
        for (int t = 0; t < 4; ++t)
            bx[t] = ld_b_frag(Bsh + (t * 16) * 40, 40, lane);
#pragma unroll
        for (int t = 0; t < 4; ++t) acc[t] = wmma_bf16(a, bx[t], acc[t]);
        if (hasNext) KB_STORE(cur ^ 1);
        __syncthreads();
        cur ^= 1;
    }
#undef KB_LOAD
#undef KB_STORE

    // Scatter mi/mo accumulators into h0 (cols 0..63 mi, 64..127 mo)
    {
        int cbase = (wave < 4) ? 0 : 64;
#pragma unroll
        for (int r = 0; r < 8; ++r) {
            int rowm = r + ((lane < 16) ? 0 : 8);
#pragma unroll
            for (int t = 0; t < 4; ++t)
                h0[(strip * 16 + rowm) * 200 + cbase + t * 16 + (lane & 15)] =
                    f2bf(acc[t][r]);
        }
    }
    // X into h0 cols 128..191 (packed column pairs, b64 loads)
#pragma unroll
    for (int j = 0; j < 8; ++j) {
        int  row = (tid >> 5) + 8 * j;
        int  c0  = (tid & 31) * 2;
        long gx  = (long)(b * NN + n0 + row) * DD + c0;
        v2f  xv  = *(const v2f*)&X[gx];
        *(unsigned*)&h0[row * 200 + 128 + c0] = pack2(xv[0], xv[1]);
    }
    __syncthreads();

    if (wave < 4) {  // 4 waves own the 4 row-strips; strip-local, no barriers
        mlp_layer(h0, 200, 192, Wf0, 200, b0, g0, bb0, h1, 72, nullptr, lane, strip);
        mlp_layer(h1, 72, 64, Wf1, 72, b1, g1, bb1, h1, 72, nullptr, lane, strip);
        mlp_layer(h1, 72, 64, Wf2, 72, b2, g2, bb2, h1, 72, nullptr, lane, strip);
        mlp_layer(h1, 72, 64, Wf3, 72, b3, g3, bb3, nullptr, 0,
                  out + (long)(b * NN + n0) * DD, lane, strip);
    }
}

// ---------------------------------------------------------------------------
extern "C" void kernel_launch(void* const* d_in, const int* in_sizes, int n_in,
                              void* d_out, int out_size, void* d_ws,
                              size_t ws_size, hipStream_t stream) {
    const float* X   = (const float*)d_in[0];
    const float* ev  = (const float*)d_in[1];
    const float* Ri  = (const float*)d_in[2];
    const float* Ro  = (const float*)d_in[3];
    const float* W0  = (const float*)d_in[4];
    const float* b0  = (const float*)d_in[5];
    const float* g0  = (const float*)d_in[6];
    const float* bb0 = (const float*)d_in[7];
    const float* W1  = (const float*)d_in[8];
    const float* b1  = (const float*)d_in[9];
    const float* g1  = (const float*)d_in[10];
    const float* bb1 = (const float*)d_in[11];
    const float* W2  = (const float*)d_in[12];
    const float* b2  = (const float*)d_in[13];
    const float* g2  = (const float*)d_in[14];
    const float* bb2 = (const float*)d_in[15];
    const float* W3  = (const float*)d_in[16];
    const float* b3  = (const float*)d_in[17];
    const float* g3  = (const float*)d_in[18];
    const float* bb3 = (const float*)d_in[19];

    unsigned short* boT  = (unsigned short*)d_ws;        // 4 MB bf16 [b][d][e]
    unsigned short* biT  = boT + (size_t)BB * EE * DD;   // 4 MB bf16 [b][d][e]
    unsigned short* XbfT = biT + (size_t)BB * EE * DD;   // 1 MB bf16 [b][d][n]

    k_xpack<<<dim3(NN / 64, BB), 256, 0, stream>>>(X, XbfT);
    k_bobi<<<dim3(EE / 64, BB), 256, 0, stream>>>(XbfT, Ri, Ro, boT, biT);
    k_mimo_mlp<<<dim3(NN / 64, BB), 256, 0, stream>>>(
        X, ev, Ri, Ro, boT, biT, W0, b0, g0, bb0, W1, b1, g1, bb1, W2, b2, g2,
        bb2, W3, b3, g3, bb3, (float*)d_out);
}
